// RotaryAttention_21225728377470
// MI455X (gfx1250) — matrix operand order repaired and checked by reading, not compiled, not run on hardware
//
#include <hip/hip_runtime.h>
#include <hip/hip_bf16.h>

// ---------------------------------------------------------------------------
// CDNA5 (gfx1250) rotary attention: bf16 WMMA GEMMs + flash attention + TDM.
// ---------------------------------------------------------------------------

typedef __attribute__((ext_vector_type(16))) __bf16 v16bf;
typedef __attribute__((ext_vector_type(8)))  __bf16 v8bf;
typedef __attribute__((ext_vector_type(8)))  float  v8f;
typedef unsigned int u32x4 __attribute__((ext_vector_type(4)));
typedef int          i32x4 __attribute__((ext_vector_type(4)));
typedef int          i32x8 __attribute__((ext_vector_type(8)));

#if defined(__has_builtin)
#if __has_builtin(__builtin_amdgcn_tensor_load_to_lds) && \
    __has_builtin(__builtin_amdgcn_s_wait_tensorcnt)
#define USE_TDM 1
#endif
#endif
#ifndef USE_TDM
#define USE_TDM 0
#endif

__device__ __forceinline__ v8f wmma_bf16(const v16bf& a, const v16bf& b, const v8f& c) {
    return __builtin_amdgcn_wmma_f32_16x16x32_bf16(false, a, false, b, (short)0, c, false, false);
}

__device__ __forceinline__ v16bf ld_frag(const __bf16* p0, const __bf16* p1) {
    // Build a 16-element bf16 fragment from two aligned 16-byte chunks.
    v8bf x0 = *(const v8bf*)p0;
    v8bf x1 = *(const v8bf*)p1;
    v16bf r;
    #pragma unroll
    for (int e = 0; e < 8; ++e) { r[e] = x0[e]; r[8 + e] = x1[e]; }
    return r;
}

#if USE_TDM
// Tensor Data Mover: async copy of a 2-D bf16 tile (32 rows x 128 elems,
// global row stride 256 elems) into LDS with 16B row padding (LDS pitch 272B).
// D# layout per CDNA5 ISA.
__device__ __forceinline__ void tdm_load_k_tile(unsigned int lds_addr,
                                                unsigned long long gaddr)
{
    u32x4 g0;
    g0[0] = 1u;                                   // count=1, user descriptor
    g0[1] = lds_addr;                             // LDS byte address
    g0[2] = (unsigned int)gaddr;                  // global_addr[31:0]
    g0[3] = (unsigned int)((gaddr >> 32) & 0x1FFFFFFu) | (2u << 30);  // addr[56:32], type=2

    i32x8 g1;
    // data_size=2B (code 1), pad_enable, pad_interval=64 dwords (code 5),
    // pad_amount=4 dwords (code 3): 256B row + 16B pad = 272B LDS pitch.
    g1[0] = (1 << 16) | (1 << 20) | (5 << 22) | (3 << 25);
    g1[1] = (128 << 16);                          // tensor_dim0 = 128  (bits 63:48)
    g1[2] = (32 << 16);                           // tensor_dim1 = 32   (bits 111:80 lo16)
    g1[3] = (128 << 16);                          // tile_dim0  = 128   (bits 127:112)
    g1[4] = 32;                                   // tile_dim1  = 32    (bits 143:128)
    g1[5] = 256;                                  // tensor_dim0_stride = 256 elems
    g1[6] = 0;
    g1[7] = 0;

    i32x4 gz = {0, 0, 0, 0};
#if __clang_major__ >= 23
    i32x8 gz8 = {0, 0, 0, 0, 0, 0, 0, 0};
    __builtin_amdgcn_tensor_load_to_lds(g0, g1, gz, gz, gz8, 0);
#else
    __builtin_amdgcn_tensor_load_to_lds(g0, g1, gz, gz, 0);
#endif
}
#endif

// ---------------------------------------------------------------------------
// Generic tiled GEMM: C[M,N] = A[M,K] @ B[K,N] + bias, double-buffered LDS.
// A is fp32 or bf16 (template); operands converted to bf16 while staging into
// LDS; compute via v_wmma_f32_16x16x32_bf16; C is bf16 or fp32.
// Block = 256 threads = 8 waves in a 4(M) x 2(N) grid; block tile 128x128.
// ---------------------------------------------------------------------------
template<int A_IS_BF16, int OUT_IS_F32>
__global__ __launch_bounds__(256)
void gemm_wmma(const void* __restrict__ Av, const float* __restrict__ B,
               const float* __restrict__ bias, void* __restrict__ Cv,
               int M, int N, int K, int lda, int ldb, int ldc)
{
    __shared__ __align__(16) __bf16 As[2][128][40];   // [m][k], pitch 40 (80B)
    __shared__ __align__(16) __bf16 Bs[2][128][40];   // [n][k] (transposed tile)

    const int tid  = threadIdx.x;
    const int lane = tid & 31;
    const int wave = tid >> 5;
    const int wm   = wave & 3;     // wave row (M)
    const int wn   = wave >> 2;    // wave col (N)
    const int half = lane >> 4;
    const int l16  = lane & 15;
    const int m0   = blockIdx.y * 128;
    const int n0   = blockIdx.x * 128;

    const float*  Af = (const float*)Av;
    const __bf16* Ab = (const __bf16*)Av;

    v8f zero = {};
    v8f acc[2][4];
    #pragma unroll
    for (int mi = 0; mi < 2; ++mi)
        #pragma unroll
        for (int ni = 0; ni < 4; ++ni) acc[mi][ni] = zero;

    float ar[16], br[16];   // per-thread staging registers (16 A + 16 B elems)

    auto global_load = [&](int k0) {
        #pragma unroll
        for (int i = 0; i < 16; ++i) {
            int e = tid + i * 256;
            int mm = e >> 5, kk = e & 31;
            if constexpr (A_IS_BF16) ar[i] = (float)Ab[(size_t)(m0 + mm) * lda + k0 + kk];
            else                     ar[i] = Af[(size_t)(m0 + mm) * lda + k0 + kk];
        }
        #pragma unroll
        for (int i = 0; i < 16; ++i) {
            int e = tid + i * 256;
            int kk = e >> 7, nn = e & 127;
            br[i] = B[(size_t)(k0 + kk) * ldb + n0 + nn];
        }
    };
    auto lds_store = [&](int buf) {
        #pragma unroll
        for (int i = 0; i < 16; ++i) {
            int e = tid + i * 256;
            As[buf][e >> 5][e & 31] = (__bf16)ar[i];
        }
        #pragma unroll
        for (int i = 0; i < 16; ++i) {
            int e = tid + i * 256;
            Bs[buf][e & 127][e >> 7] = (__bf16)br[i];
        }
    };

    const int ntiles = K >> 5;
    global_load(0);
    lds_store(0);
    __syncthreads();

    for (int kt = 0; kt < ntiles; ++kt) {
        const int cur = kt & 1;
        if (kt + 1 < ntiles) global_load((kt + 1) << 5);   // overlap with compute
        if (kt + 2 < ntiles)
            __builtin_prefetch(&B[(size_t)((kt + 2) << 5) * ldb + n0 + (tid & 127)], 0, 1);

        // A fragments: lane l16 = row; halves split K per ISA 16-bit A layout.
        v16bf a[2], b[4];
        #pragma unroll
        for (int mi = 0; mi < 2; ++mi) {
            const __bf16* row = &As[cur][wm * 32 + mi * 16 + l16][0];
            a[mi] = ld_frag(row + half * 8, row + 16 + half * 8);
        }
        // B fragments: lane l16 = column; halves split K like the A layout.
        #pragma unroll
        for (int ni = 0; ni < 4; ++ni) {
            const __bf16* row = &Bs[cur][wn * 64 + ni * 16 + l16][0];
            b[ni] = ld_frag(row + half * 8, row + 16 + half * 8);
        }
        #pragma unroll
        for (int mi = 0; mi < 2; ++mi)
            #pragma unroll
            for (int ni = 0; ni < 4; ++ni)
                acc[mi][ni] = wmma_bf16(a[mi], b[ni], acc[mi][ni]);

        if (kt + 1 < ntiles) lds_store(cur ^ 1);
        __syncthreads();
    }

    // Epilogue: C-layout VGPR j -> row j (lanes 0-15) / j+8 (lanes 16-31).
    #pragma unroll
    for (int mi = 0; mi < 2; ++mi) {
        #pragma unroll
        for (int ni = 0; ni < 4; ++ni) {
            int col = n0 + wn * 64 + ni * 16 + l16;
            float bv = bias[col];
            #pragma unroll
            for (int j = 0; j < 8; ++j) {
                int row = m0 + wm * 32 + mi * 16 + j + half * 8;
                float v = acc[mi][ni][j] + bv;
                if constexpr (OUT_IS_F32) ((float*)Cv)[(size_t)row * ldc + col] = v;
                else                      ((__bf16*)Cv)[(size_t)row * ldc + col] = (__bf16)v;
            }
        }
    }
}

// ---------------------------------------------------------------------------
// RoPE applied in place to a bf16 [4096, nh, 128] tensor. One thread per
// (token, head, d<64) pair; reads both rotation partners, writes both.
// ---------------------------------------------------------------------------
__global__ __launch_bounds__(256)
void rope_kernel(__bf16* __restrict__ buf, int nh, int total)
{
    int idx = blockIdx.x * blockDim.x + threadIdx.x;
    if (idx >= total) return;
    int d  = idx & 63;
    int th = idx >> 6;          // token*nh + head
    int tok = th / nh;
    int pos = tok & 1023;       // position within sequence
    __bf16* p = buf + (size_t)th * 128;
    float xa = (float)p[d];
    float xb = (float)p[d + 64];
    // inv_freq = 10000^(-d/64)
    float inv = __expf(-(float)d * (9.210340371976184f / 64.0f));
    float f = (float)pos * inv;
    float sf, cf;
    __sincosf(f, &sf, &cf);
    p[d]      = (__bf16)(xa * cf - xb * sf);
    p[d + 64] = (__bf16)(xb * cf + xa * sf);
}

// ---------------------------------------------------------------------------
// Flash attention (GQA): grid.x = (b,g,r) flat (64), grid.y = query tile (16).
// Block = 128 threads = 4 waves; each wave owns 16 query rows x 128 dims.
// K tile staged by the Tensor Data Mover (async DMA + s_wait_tensorcnt) when
// available; V tile transposed by VALU. Online softmax, bf16 WMMA throughout.
// ---------------------------------------------------------------------------
__global__ __launch_bounds__(128)
void flash_attn(const __bf16* __restrict__ Q, const __bf16* __restrict__ Kb,
                const __bf16* __restrict__ Vb, __bf16* __restrict__ AO)
{
    __shared__ __align__(16) __bf16 Ks[32][136];     // K tile, row-major [t][d], pitch 272B
    __shared__ __align__(16) __bf16 Vt[128][40];     // V tile transposed [d][t]
    __shared__ __align__(16) __bf16 Pst[4][16][40];  // per-wave P staging [m][t]

    const int tid  = threadIdx.x;
    const int lane = tid & 31;
    const int wave = tid >> 5;
    const int half = lane >> 4;
    const int l16  = lane & 15;

    const int bh   = blockIdx.x;           // 0..63
    const int bIdx = bh >> 4;
    const int g    = (bh >> 3) & 1;
    const int r    = bh & 7;
    const int head = g * 8 + r;
    const int qrow0 = blockIdx.y * 64 + wave * 16;

    // Preload Q fragments for this wave's 16 rows (4 chunks of K=32 over hd=128).
    v16bf qf[4];
    const __bf16* qbase = Q + ((size_t)(bIdx * 1024 + qrow0 + l16) * 16 + head) * 128;
    #pragma unroll
    for (int c = 0; c < 4; ++c)
        qf[c] = ld_frag(qbase + c * 32 + half * 8, qbase + c * 32 + 16 + half * 8);

    v8f zero = {};
    v8f o[8];
    #pragma unroll
    for (int i = 0; i < 8; ++i) o[i] = zero;
    float m_run[8], l_run[8];
    #pragma unroll
    for (int j = 0; j < 8; ++j) { m_run[j] = -1e30f; l_run[j] = 0.0f; }

    const float scale = 0.08838834764831845f;   // 1/sqrt(128)

    for (int t0 = 0; t0 < 1024; t0 += 32) {
        __syncthreads();
#if USE_TDM
        // Async DMA of the K tile (TDM handles the strided rows + LDS padding).
        if (wave == 0) {
            tdm_load_k_tile(
                (unsigned int)(size_t)&Ks[0][0],
                (unsigned long long)(size_t)(Kb + ((size_t)(bIdx * 1024 + t0) * 2 + g) * 128));
        }
        // Meanwhile all waves transpose the V tile with VALU stores.
        for (int e = tid; e < 32 * 128; e += 128) {
            int t = e >> 7, dd = e & 127;
            Vt[dd][t] = Vb[((size_t)(bIdx * 1024 + t0 + t) * 2 + g) * 128 + dd];
        }
        if (wave == 0) __builtin_amdgcn_s_wait_tensorcnt(0);
#else
        for (int e = tid; e < 32 * 128; e += 128) {
            int t = e >> 7, dd = e & 127;
            size_t src = ((size_t)(bIdx * 1024 + t0 + t) * 2 + g) * 128 + dd;
            Ks[t][dd] = Kb[src];
            Vt[dd][t] = Vb[src];
        }
#endif
        __syncthreads();

        // Batch all 8 K^T fragments, then the two interleaved WMMA chains.
        v16bf kf[8];
        #pragma unroll
        for (int c = 0; c < 4; ++c) {
            const __bf16* kr0 = &Ks[l16][c * 32 + half * 8];
            const __bf16* kr1 = &Ks[16 + l16][c * 32 + half * 8];
            kf[c]     = ld_frag(kr0, kr0 + 16);
            kf[4 + c] = ld_frag(kr1, kr1 + 16);
        }
        v8f s0 = zero, s1 = zero;
        #pragma unroll
        for (int c = 0; c < 4; ++c) {
            s0 = wmma_bf16(qf[c], kf[c], s0);
            s1 = wmma_bf16(qf[c], kf[4 + c], s1);
        }

        // Online softmax. Row of VGPR j lives in one 16-lane half; xor masks
        // 1..8 reduce within that half (wave32).
        #pragma unroll
        for (int j = 0; j < 8; ++j) {
            float a0 = s0[j] * scale;
            float a1 = s1[j] * scale;
            float rm = fmaxf(a0, a1);
            #pragma unroll
            for (int mk = 1; mk <= 8; mk <<= 1) rm = fmaxf(rm, __shfl_xor(rm, mk, 32));
            float mn = fmaxf(m_run[j], rm);
            float sc = __expf(m_run[j] - mn);
            m_run[j] = mn;
            float p0 = __expf(a0 - mn);
            float p1 = __expf(a1 - mn);
            float rs = p0 + p1;
            #pragma unroll
            for (int mk = 1; mk <= 8; mk <<= 1) rs += __shfl_xor(rs, mk, 32);
            l_run[j] = l_run[j] * sc + rs;
            #pragma unroll
            for (int di = 0; di < 8; ++di) o[di][j] *= sc;
            int mrow = j + half * 8;
            Pst[wave][mrow][l16]      = (__bf16)p0;
            Pst[wave][mrow][16 + l16] = (__bf16)p1;
        }
        // Same-wave LDS RAW: wait for the P stores before fragment reload.
        asm volatile("s_wait_dscnt 0" ::: "memory");

        // P back in A-fragment layout; batch all 8 V fragments, then 8 WMMAs.
        const __bf16* pr = &Pst[wave][l16][0];
        v16bf pf = ld_frag(pr + half * 8, pr + 16 + half * 8);
        v16bf vf[8];
        #pragma unroll
        for (int di = 0; di < 8; ++di) {
            const __bf16* vr = &Vt[di * 16 + l16][half * 8];
            vf[di] = ld_frag(vr, vr + 16);
        }
        #pragma unroll
        for (int di = 0; di < 8; ++di)
            o[di] = wmma_bf16(pf, vf[di], o[di]);
    }

    // Epilogue: normalize and scatter to attn-out [b, s, head, 128] (bf16).
    #pragma unroll
    for (int di = 0; di < 8; ++di) {
        #pragma unroll
        for (int j = 0; j < 8; ++j) {
            int row = qrow0 + j + half * 8;
            int dd  = di * 16 + l16;
            float v = o[di][j] / l_run[j];
            AO[((size_t)(bIdx * 1024 + row) * 16 + head) * 128 + dd] = (__bf16)v;
        }
    }
}

// ---------------------------------------------------------------------------
// Host-side orchestration.
// ---------------------------------------------------------------------------
extern "C" void kernel_launch(void* const* d_in, const int* in_sizes, int n_in,
                              void* d_out, int out_size, void* d_ws, size_t ws_size,
                              hipStream_t stream)
{
    const float* x  = (const float*)d_in[0];
    const float* Wq = (const float*)d_in[1];
    const float* bq = (const float*)d_in[2];
    const float* Wk = (const float*)d_in[3];
    const float* bk = (const float*)d_in[4];
    const float* Wv = (const float*)d_in[5];
    const float* bv = (const float*)d_in[6];
    const float* Wo = (const float*)d_in[7];
    const float* bo = (const float*)d_in[8];
    float* out = (float*)d_out;

    // Workspace layout (bf16 intermediates): Q 16MB, K 2MB, V 2MB, AO 16MB.
    char* ws = (char*)d_ws;
    __bf16* Qb = (__bf16*)(ws);
    __bf16* Kb = (__bf16*)(ws + (size_t)16777216);
    __bf16* Vb = (__bf16*)(ws + (size_t)16777216 + 2097152);
    __bf16* AO = (__bf16*)(ws + (size_t)16777216 + 2 * 2097152);

    dim3 blk256(256);

    // Q/K/V projections (fp32 in -> bf16 out, bias fused).
    gemm_wmma<0, 0><<<dim3(16, 32), blk256, 0, stream>>>(x, Wq, bq, Qb,
        4096, 2048, 2048, 2048, 2048, 2048);
    gemm_wmma<0, 0><<<dim3(2, 32), blk256, 0, stream>>>(x, Wk, bk, Kb,
        4096, 256, 2048, 2048, 256, 256);
    gemm_wmma<0, 0><<<dim3(2, 32), blk256, 0, stream>>>(x, Wv, bv, Vb,
        4096, 256, 2048, 2048, 256, 256);

    // RoPE in place on Q and K.
    int nq = 4096 * 16 * 64;
    rope_kernel<<<(nq + 255) / 256, 256, 0, stream>>>(Qb, 16, nq);
    int nk = 4096 * 2 * 64;
    rope_kernel<<<(nk + 255) / 256, 256, 0, stream>>>(Kb, 2, nk);

    // Flash attention: 64 (b,g,r) heads x 16 query tiles of 64 rows.
    flash_attn<<<dim3(64, 16), dim3(128), 0, stream>>>(Qb, Kb, Vb, AO);

    // Output projection (bf16 A, fp32 out + bias).
    gemm_wmma<1, 1><<<dim3(16, 32), blk256, 0, stream>>>(AO, Wo, bo, out,
        4096, 2048, 2048, 2048, 2048, 2048);
}
